// Transformer_66039417143602
// MI455X (gfx1250) — matrix operand order repaired and checked by reading, not compiled, not run on hardware
//
#include <hip/hip_runtime.h>
#include <hip/hip_bf16.h>

// ---------------------------------------------------------------------------
// MI455X (gfx1250) transformer forward: all matmuls via v_wmma_f32_16x16x32_f16
//  - async global->LDS staging (ASYNCcnt) when the toolchain exposes it
//  - permlane16/permlanex16 cross-lane reductions (VALU) instead of ds_bpermute
// ---------------------------------------------------------------------------

typedef _Float16 v8h  __attribute__((ext_vector_type(8)));
typedef _Float16 v16h __attribute__((ext_vector_type(16)));
typedef float    v8f  __attribute__((ext_vector_type(8)));

static constexpr int NB     = 4;     // batch
static constexpr int SEQ    = 1024;  // seq len
static constexpr int EMB    = 1024;  // embed dim
static constexpr int NHEAD  = 16;
static constexpr int HDIM   = 64;
static constexpr int QKVD   = 1024;
static constexpr int MLPD   = 4096;
static constexpr int NLAYER = 4;

#define SHUF16(lo, hi) __builtin_shufflevector((lo), (hi), 0,1,2,3,4,5,6,7,8,9,10,11,12,13,14,15)

#if __has_builtin(__builtin_amdgcn_global_load_async_to_lds_b128)
#define USE_ASYNC_LDS 1
#endif
#if __has_builtin(__builtin_amdgcn_permlane16)
#define USE_PERMLANE 1
#endif

// exact pointee type of the async builtin's parameters: 16-byte int vector
typedef int v4i_vs __attribute__((vector_size(16)));
typedef __attribute__((address_space(1))) v4i_vs GV4;  // global (__device__)
typedef __attribute__((address_space(3))) v4i_vs LV4;  // LDS (__shared__)

__device__ __forceinline__ v8f wmma_f16(v16h a, v16h b, v8f c) {
  // D = A(16x32,f16) x B(32x16,f16) + C(16x16,f32)
  return __builtin_amdgcn_wmma_f32_16x16x32_f16(false, a, false, b, (short)0, c, false, false);
}

// 16-byte global -> LDS copy (async engine when available)
__device__ __forceinline__ void cp16_async(_Float16* l, const _Float16* g) {
#ifdef USE_ASYNC_LDS
  __builtin_amdgcn_global_load_async_to_lds_b128((GV4*)g, (LV4*)l, 0, 0);
#else
  *(v8h*)l = *(const v8h*)g;
#endif
}

__device__ __forceinline__ void wait_async() {
#ifdef USE_ASYNC_LDS
#if __has_builtin(__builtin_amdgcn_s_wait_asynccnt)
  __builtin_amdgcn_s_wait_asynccnt(0);
#else
  asm volatile("s_wait_asynccnt 0" ::: "memory");
#endif
#endif
}

// xor-lane exchange; m in {1,2,4,8} stays within a 16-lane row (v_permlane16),
// m == 16 swaps rows (v_permlanex16). Falls back to __shfl_xor.
__device__ __forceinline__ float row_xor(float v, int m) {
#ifdef USE_PERMLANE
  unsigned int s0, s1;
  unsigned int u = __builtin_bit_cast(unsigned int, v);
  if (m == 16) {
    u = __builtin_amdgcn_permlanex16(u, u, 0x76543210u, 0xFEDCBA98u, false, false);
  } else {
    switch (m) {
      case 1:  s0 = 0x67452301u; s1 = 0xEFCDAB89u; break;
      case 2:  s0 = 0x54761032u; s1 = 0xDCFE98BAu; break;
      case 4:  s0 = 0x32107654u; s1 = 0xBA98FEDCu; break;
      default: s0 = 0xFEDCBA98u; s1 = 0x76543210u; break;  // m == 8
    }
    u = __builtin_amdgcn_permlane16(u, u, s0, s1, false, false);
  }
  return __builtin_bit_cast(float, u);
#else
  return __shfl_xor(v, m, 32);
#endif
}

// -------------------------------------------------------------- cast f32->f16
__global__ __launch_bounds__(256) void cast_f32_f16(const float* __restrict__ in,
                                                    _Float16* __restrict__ out, int n) {
  int i = blockIdx.x * 256 + threadIdx.x;
  if (i < n) out[i] = (_Float16)in[i];
}

// ------------------------------------------- weight convert: Wt[n][k] = W[k][n]
__global__ __launch_bounds__(256) void convert_wt(const float* __restrict__ W,
                                                  _Float16* __restrict__ Wt,
                                                  int K, int N) {
  int idx = blockIdx.x * 256 + threadIdx.x;
  if (idx < N * K) {
    int n = idx / K, k = idx - n * K;
    Wt[idx] = (_Float16)W[(size_t)k * N + n];
  }
}

// ---------------------------------------------------------------- tiled GEMM
// C[M][N] = A16[M][K] * Bt16[N][K]^T ; out = (acc + bias[n]) * scale ; opt ReLU
template <bool OUT_F16, bool RELU>
__global__ __launch_bounds__(256)
void gemm_kernel(const _Float16* __restrict__ A, const _Float16* __restrict__ Bt,
                 const float* __restrict__ bias, float scale,
                 void* __restrict__ out, int M, int N, int K) {
  constexpr int BM = 128, BN = 128, BK = 32, LDT = BK + 8;  // pad: rows 16B-aligned
  __shared__ _Float16 Al[BM * LDT];
  __shared__ _Float16 Bl[BN * LDT];

  const int tid  = threadIdx.x;
  const int wid  = tid >> 5;
  const int lane = tid & 31;
  const int sub  = lane >> 4;   // half-wave (0/1)
  const int l16  = lane & 15;
  const int m0   = blockIdx.x * BM;
  const int n0   = blockIdx.y * BN;
  const int wm   = (wid >> 1) * 32;  // wave row offset in tile
  const int wn   = (wid & 1) * 64;   // wave col offset in tile

  const v8f vzero = {};
  v8f acc[2][4];
#pragma unroll
  for (int i = 0; i < 2; ++i)
#pragma unroll
    for (int j = 0; j < 4; ++j) acc[i][j] = vzero;

  const int lr = tid >> 1;          // 0..127 : tile row loaded by this thread
  const int lc = (tid & 1) * 16;    // 0 / 16 : k-offset loaded by this thread

  for (int k0 = 0; k0 < K; k0 += BK) {
    __syncthreads();  // previous iteration's fragment reads done
    const _Float16* ag = A  + (size_t)(m0 + lr) * K + k0 + lc;
    const _Float16* bg = Bt + (size_t)(n0 + lr) * K + k0 + lc;
    cp16_async(&Al[lr * LDT + lc],     ag);
    cp16_async(&Al[lr * LDT + lc + 8], ag + 8);
    cp16_async(&Bl[lr * LDT + lc],     bg);
    cp16_async(&Bl[lr * LDT + lc + 8], bg + 8);
    __builtin_prefetch(ag + BK, 0, 1);   // -> global_prefetch_b8
    __builtin_prefetch(bg + BK, 0, 1);
    wait_async();
    __syncthreads();

    // A fragments: lane row = l16; halves j -> k = (j&7) + (j>=8)*16 + sub*8
    v16h afrag[2];
#pragma unroll
    for (int mt = 0; mt < 2; ++mt) {
      const _Float16* ap = &Al[(wm + mt * 16 + l16) * LDT + sub * 8];
      v8h lo = *(const v8h*)ap;
      v8h hi = *(const v8h*)(ap + 16);
      afrag[mt] = SHUF16(lo, hi);
    }
    // B fragments: lane col = l16; halves j -> k = (j&7) + (j>=8)*16 + sub*8
#pragma unroll
    for (int nt = 0; nt < 4; ++nt) {
      const _Float16* bp = &Bl[(wn + nt * 16 + l16) * LDT + sub * 8];
      v8h lo = *(const v8h*)bp;
      v8h hi = *(const v8h*)(bp + 16);
      v16h bfrag = SHUF16(lo, hi);
#pragma unroll
      for (int mt = 0; mt < 2; ++mt) acc[mt][nt] = wmma_f16(afrag[mt], bfrag, acc[mt][nt]);
    }
  }

  // epilogue: C layout — lane col = l16, VGPR r -> row = r + 8*sub
#pragma unroll
  for (int mt = 0; mt < 2; ++mt)
#pragma unroll
    for (int nt = 0; nt < 4; ++nt) {
      const int col = n0 + wn + nt * 16 + l16;
      const float bv = bias[col];
#pragma unroll
      for (int r = 0; r < 8; ++r) {
        const int row = m0 + wm + mt * 16 + sub * 8 + r;
        float v = (acc[mt][nt][r] + bv) * scale;
        if (RELU) v = v > 0.f ? v : 0.f;
        if constexpr (OUT_F16)
          ((_Float16*)out)[(size_t)row * N + col] = (_Float16)v;
        else
          ((float*)out)[(size_t)row * N + col] = v;
      }
    }
}

// --------------------------------------------------------- flash attention
// q pre-scaled by 1/sqrt(hd). Layouts: [B][L][H*64].
__global__ __launch_bounds__(256)
void attention_kernel(const _Float16* __restrict__ q, const _Float16* __restrict__ k,
                      const _Float16* __restrict__ v, _Float16* __restrict__ o) {
  constexpr int KCH = 32;           // kv chunk
  constexpr int LDK = HDIM + 8;     // Klds stride (halves)
  constexpr int LDV = KCH + 8;      // Vlds stride
  constexpr int PLD = 40;           // Plds stride (16x32 tile + pad)
  __shared__ _Float16 Klds[KCH * LDK];    // K chunk row-major [32][64]
  __shared__ _Float16 Vlds[HDIM * LDV];   // V chunk transposed [64][32]
  __shared__ _Float16 Plds[8 * 16 * PLD]; // per-wave P scratch

  const int tid  = threadIdx.x;
  const int wid  = tid >> 5;
  const int lane = tid & 31;
  const int sub  = lane >> 4;
  const int l16  = lane & 15;

  const int nqb  = SEQ / 128;
  const int qblk = blockIdx.x % nqb;
  const int h    = (blockIdx.x / nqb) % NHEAD;
  const int b    = blockIdx.x / (nqb * NHEAD);
  const size_t base = (size_t)b * SEQ * EMB + (size_t)h * HDIM;

  // Q fragments (A layout), rows = qblk*128 + wid*16 + l16, d split 0-31 / 32-63
  const int qrow = qblk * 128 + wid * 16 + l16;
  v16h qf[2];
#pragma unroll
  for (int s = 0; s < 2; ++s) {
    const _Float16* qp = q + base + (size_t)qrow * EMB + s * 32 + sub * 8;
    v8h lo = *(const v8h*)qp;
    v8h hi = *(const v8h*)(qp + 16);
    qf[s] = SHUF16(lo, hi);
  }

  const v8f vzero = {};
  v8f Oacc[4];
#pragma unroll
  for (int t = 0; t < 4; ++t) Oacc[t] = vzero;
  float mrun[8], srun[8];
#pragma unroll
  for (int r = 0; r < 8; ++r) { mrun[r] = -1e30f; srun[r] = 0.f; }

  const int pbase = wid * 16 * PLD;

  for (int c = 0; c < SEQ; c += KCH) {
    __syncthreads();
    {  // cooperative stage: K row-major (async), V transposed (2B scatter)
      const int r   = tid >> 3;          // 0..31
      const int col = (tid & 7) * 8;     // 0..56
      cp16_async(&Klds[r * LDK + col], k + base + (size_t)(c + r) * EMB + col);
      v8h vv = *(const v8h*)(v + base + (size_t)(c + r) * EMB + col);
#pragma unroll
      for (int j = 0; j < 8; ++j) Vlds[(col + j) * LDV + r] = vv[j];
    }
    wait_async();
    __syncthreads();

    // S = Q K^T : two 16x16 tiles (cols 0-15, 16-31), each sums over d in 2 steps
    v8f st[2] = {vzero, vzero};
#pragma unroll
    for (int t = 0; t < 2; ++t) {
#pragma unroll
      for (int s = 0; s < 2; ++s) {
        const _Float16* bp = &Klds[(t * 16 + l16) * LDK + s * 32 + sub * 8];
        v8h lo = *(const v8h*)bp;
        v8h hi = *(const v8h*)(bp + 16);
        st[t] = wmma_f16(qf[s], SHUF16(lo, hi), st[t]);
      }
    }

    // online softmax over 32 cols (row stats live in 16-lane half-waves)
#pragma unroll
    for (int r = 0; r < 8; ++r) {
      float mx = fmaxf(st[0][r], st[1][r]);
#pragma unroll
      for (int m = 8; m >= 1; m >>= 1) mx = fmaxf(mx, row_xor(mx, m));
      const float nm   = fmaxf(mrun[r], mx);
      const float corr = __expf(mrun[r] - nm);
      const float p0   = __expf(st[0][r] - nm);
      const float p1   = __expf(st[1][r] - nm);
      float rs = p0 + p1;
#pragma unroll
      for (int m = 8; m >= 1; m >>= 1) rs += row_xor(rs, m);
      srun[r] = srun[r] * corr + rs;
      mrun[r] = nm;
#pragma unroll
      for (int t = 0; t < 4; ++t) Oacc[t][r] *= corr;
      // C-layout store of P tile into per-wave scratch
      Plds[pbase + (r + 8 * sub) * PLD + l16]      = (_Float16)p0;
      Plds[pbase + (r + 8 * sub) * PLD + 16 + l16] = (_Float16)p1;
    }
    asm volatile("" ::: "memory");  // per-wave LDS ops are in-order; block reordering

    // reload P as A fragment (16x32)
    const _Float16* pp = &Plds[pbase + l16 * PLD + sub * 8];
    v8h plo = *(const v8h*)pp;
    v8h phi = *(const v8h*)(pp + 16);
    v16h pf = SHUF16(plo, phi);

    // O += P * V : 4 tiles of 16 d-columns
#pragma unroll
    for (int t = 0; t < 4; ++t) {
      const _Float16* bp = &Vlds[(t * 16 + l16) * LDV + sub * 8];
      v8h lo = *(const v8h*)bp;
      v8h hi = *(const v8h*)(bp + 16);
      Oacc[t] = wmma_f16(pf, SHUF16(lo, hi), Oacc[t]);
    }
  }

  // normalize + store (C layout rows)
#pragma unroll
  for (int t = 0; t < 4; ++t)
#pragma unroll
    for (int r = 0; r < 8; ++r) {
      const int row = qblk * 128 + wid * 16 + 8 * sub + r;
      const int col = t * 16 + l16;
      o[base + (size_t)row * EMB + col] = (_Float16)(Oacc[t][r] / srun[r]);
    }
}

// ------------------------------------------------- residual add + LayerNorm
__global__ __launch_bounds__(256)
void add_ln_kernel(const float* __restrict__ resid, const float* __restrict__ y,
                   const float* __restrict__ g, const float* __restrict__ bt,
                   float* __restrict__ out32, _Float16* __restrict__ out16) {
  const int row = blockIdx.x, tid = threadIdx.x;
  const int lane = tid & 31, wid = tid >> 5;
  const size_t base = (size_t)row * EMB;
  __shared__ float red1[8], red2[8];

  float loc[4];
  float s = 0.f;
#pragma unroll
  for (int i = 0; i < 4; ++i) {
    const int c = tid + i * 256;
    loc[i] = resid[base + c] + y[base + c];
    s += loc[i];
  }
#pragma unroll
  for (int m = 16; m >= 1; m >>= 1) s += row_xor(s, m);
  if (lane == 0) red1[wid] = s;
  __syncthreads();
  float tot = 0.f;
#pragma unroll
  for (int w = 0; w < 8; ++w) tot += red1[w];
  const float mean = tot * (1.f / EMB);

  float vs = 0.f;
#pragma unroll
  for (int i = 0; i < 4; ++i) { const float d = loc[i] - mean; vs += d * d; }
#pragma unroll
  for (int m = 16; m >= 1; m >>= 1) vs += row_xor(vs, m);
  if (lane == 0) red2[wid] = vs;
  __syncthreads();
  float vtot = 0.f;
#pragma unroll
  for (int w = 0; w < 8; ++w) vtot += red2[w];
  const float rstd = rsqrtf(vtot * (1.f / EMB) + 1e-6f);

#pragma unroll
  for (int i = 0; i < 4; ++i) {
    const int c = tid + i * 256;
    const float ov = (loc[i] - mean) * rstd * g[c] + bt[c];
    out32[base + c] = ov;
    out16[base + c] = (_Float16)ov;
  }
}

// --------------------------------------------------------------------- host
extern "C" void kernel_launch(void* const* d_in, const int* in_sizes, int n_in,
                              void* d_out, int out_size, void* d_ws, size_t ws_size,
                              hipStream_t stream) {
  (void)in_sizes; (void)n_in; (void)out_size; (void)ws_size;
  const float* queries = (const float*)d_in[0];
  const float* Wq = (const float*)d_in[1];  const float* bq = (const float*)d_in[2];
  const float* Wk = (const float*)d_in[3];  const float* bk = (const float*)d_in[4];
  const float* Wv = (const float*)d_in[5];  const float* bv = (const float*)d_in[6];
  const float* Wo = (const float*)d_in[7];  const float* bo = (const float*)d_in[8];
  const float* ln1s = (const float*)d_in[9];  const float* ln1b = (const float*)d_in[10];
  const float* ln2s = (const float*)d_in[11]; const float* ln2b = (const float*)d_in[12];
  const float* W1 = (const float*)d_in[13]; const float* b1 = (const float*)d_in[14];
  const float* W2 = (const float*)d_in[15]; const float* b2 = (const float*)d_in[16];

  const int M = NB * SEQ;  // 4096 rows
  char* p = (char*)d_ws;
  auto take = [&](size_t bytes) -> void* {
    void* r = (void*)p;
    p += (bytes + 255) & ~(size_t)255;
    return r;
  };
  _Float16* x16   = (_Float16*)take((size_t)M * EMB * 2);
  float*    x32   = (float*)   take((size_t)M * EMB * 4);
  _Float16* q16   = (_Float16*)take((size_t)M * QKVD * 2);
  _Float16* k16   = (_Float16*)take((size_t)M * QKVD * 2);
  _Float16* v16   = (_Float16*)take((size_t)M * QKVD * 2);
  _Float16* upd16 = (_Float16*)take((size_t)M * QKVD * 2);
  float*    a32   = (float*)   take((size_t)M * EMB * 4);
  float*    h32   = (float*)   take((size_t)M * EMB * 4);
  _Float16* h16   = (_Float16*)take((size_t)M * EMB * 2);
  _Float16* z1    = (_Float16*)take((size_t)M * MLPD * 2);
  float*    z32   = (float*)   take((size_t)M * EMB * 4);
  _Float16* wq_t  = (_Float16*)take((size_t)EMB * QKVD * 2);
  _Float16* wk_t  = (_Float16*)take((size_t)EMB * QKVD * 2);
  _Float16* wv_t  = (_Float16*)take((size_t)EMB * QKVD * 2);
  _Float16* wo_t  = (_Float16*)take((size_t)QKVD * EMB * 2);
  _Float16* w1_t  = (_Float16*)take((size_t)EMB * MLPD * 2);
  _Float16* w2_t  = (_Float16*)take((size_t)MLPD * EMB * 2);

  cast_f32_f16<<<(M * EMB + 255) / 256, 256, 0, stream>>>(queries, x16, M * EMB);

  const dim3 gQKV(M / 128, QKVD / 128);
  const dim3 gE(M / 128, EMB / 128);
  const dim3 gMLP(M / 128, MLPD / 128);
  const int cvtSmall = (EMB * QKVD + 255) / 256;
  const int cvtBig   = (EMB * MLPD + 255) / 256;

  for (int l = 0; l < NLAYER; ++l) {
    const float* xin32 = (l == 0) ? queries : x32;
    convert_wt<<<cvtSmall, 256, 0, stream>>>(Wq + (size_t)l * EMB * QKVD, wq_t, EMB, QKVD);
    convert_wt<<<cvtSmall, 256, 0, stream>>>(Wk + (size_t)l * EMB * QKVD, wk_t, EMB, QKVD);
    convert_wt<<<cvtSmall, 256, 0, stream>>>(Wv + (size_t)l * EMB * QKVD, wv_t, EMB, QKVD);
    convert_wt<<<cvtSmall, 256, 0, stream>>>(Wo + (size_t)l * QKVD * EMB, wo_t, QKVD, EMB);
    convert_wt<<<cvtBig,   256, 0, stream>>>(W1 + (size_t)l * EMB * MLPD, w1_t, EMB, MLPD);
    convert_wt<<<cvtBig,   256, 0, stream>>>(W2 + (size_t)l * MLPD * EMB, w2_t, MLPD, EMB);

    gemm_kernel<true, false><<<gQKV, 256, 0, stream>>>(x16, wq_t, bq + (size_t)l * QKVD,
                                                       0.125f, q16, M, QKVD, EMB);
    gemm_kernel<true, false><<<gQKV, 256, 0, stream>>>(x16, wk_t, bk + (size_t)l * QKVD,
                                                       1.0f, k16, M, QKVD, EMB);
    gemm_kernel<true, false><<<gQKV, 256, 0, stream>>>(x16, wv_t, bv + (size_t)l * QKVD,
                                                       1.0f, v16, M, QKVD, EMB);
    attention_kernel<<<NB * NHEAD * (SEQ / 128), 256, 0, stream>>>(q16, k16, v16, upd16);
    gemm_kernel<false, false><<<gE, 256, 0, stream>>>(upd16, wo_t, bo + (size_t)l * EMB,
                                                      1.0f, a32, M, EMB, QKVD);
    add_ln_kernel<<<M, 256, 0, stream>>>(xin32, a32, ln1s + (size_t)l * EMB,
                                         ln1b + (size_t)l * EMB, h32, h16);
    gemm_kernel<true, true><<<gMLP, 256, 0, stream>>>(h16, w1_t, b1 + (size_t)l * MLPD,
                                                      1.0f, z1, M, MLPD, EMB);
    gemm_kernel<false, false><<<gE, 256, 0, stream>>>(z1, w2_t, b2 + (size_t)l * EMB,
                                                      1.0f, z32, M, EMB, MLPD);
    float* xout = (l == NLAYER - 1) ? (float*)d_out : x32;
    add_ln_kernel<<<M, 256, 0, stream>>>(h32, z32, ln2s + (size_t)l * EMB,
                                         ln2b + (size_t)l * EMB, xout, x16);
  }
}
